// HeteroGNNLayer_81372450390255
// MI455X (gfx1250) — matrix operand order repaired and checked by reading, not compiled, not run on hardware
//
#include <hip/hip_runtime.h>

// ---------------------------------------------------------------------------
// HeteroGNN layer for MI455X (gfx1250, wave32, WMMA).
// X[src] @ W == (X @ W)[src]  -> dense f16-WMMA pre-projection, then a light
// edge-MLP WMMA + gather + ReLU + native-f32-atomic scatter.
// Weights are pre-swizzled into WMMA B-fragment order so every fragment load
// is one aligned 32B vector load instead of 16 strided u16 loads.
// ---------------------------------------------------------------------------

typedef __attribute__((ext_vector_type(16))) _Float16 v16h;
typedef __attribute__((ext_vector_type(8)))  _Float16 v8h;
typedef __attribute__((ext_vector_type(2)))  _Float16 h2;
typedef __attribute__((ext_vector_type(8)))  float    v8f;

#define GNN_N_USER 100000
#define GNN_N_ITEM 100000
#define GNN_E      800000
#define GNN_DIN    128
#define GNN_DEDGE  32
#define GNN_DOUT   128

// ------------------- pack W[K,128] f32 -> B-fragment-major f16 -------------
// Wp[((kc*8 + g)*32 + lane)*16 + i] = f16( W[(kc*32 + (lane>>4)*16 + i)*128
//                                            + g*16 + (lane&15)] )
__global__ __launch_bounds__(256) void gnn_pack_w(const float* __restrict__ W,
                                                  _Float16* __restrict__ Wp,
                                                  int K) {
  int j = blockIdx.x * 256 + threadIdx.x;     // total K*128 elements
  if (j >= K * 128) return;
  int i    = j & 15;
  int lane = (j >> 4) & 31;
  int g    = (j >> 9) & 7;
  int kc   = j >> 12;
  int k    = kc * 32 + ((lane >> 4) << 4) + i;
  int col  = g * 16 + (lane & 15);
  Wp[j] = (_Float16)W[(size_t)k * GNN_DOUT + col];
}

// --------------------------- dense node GEMM -------------------------------
// Y[rowBase:rowBase+16, 0:128] = X_tile @ W (+ bias).  K = 128 = 4 x WMMA-K32.
// 256 threads = 8 waves; wave w owns col group w. X tile staged in LDS as f16.
__global__ __launch_bounds__(256) void gnn_node_gemm(
    const float*    __restrict__ X,    // [N,128] f32 row-major
    const _Float16* __restrict__ Wp,   // packed fragments: [4][8][32][16]
    const float*    __restrict__ bias, // [128] or nullptr
    float*          __restrict__ Y)    // [N,128] f32
{
  __shared__ __align__(16) h2 xs[16 * 68];    // [row][k/2], pad 68 -> no conflicts
  const int t       = threadIdx.x;
  const int rowBase = blockIdx.x * 16;

  // Stage + convert the 16x128 X tile once per block (coalesced float2 reads).
  const float2* X2 = (const float2*)X;
  for (int idx = t; idx < 16 * 64; idx += 256) {
    int r = idx >> 6, c2 = idx & 63;
    float2 v = X2[(size_t)(rowBase + r) * 64 + c2];
    h2 p = {(_Float16)v.x, (_Float16)v.y};
    xs[r * 68 + c2] = p;
  }
  __syncthreads();

  const int wave = t >> 5;
  const int lane = t & 31;
  const int hi   = lane >> 4;
  const int l15  = lane & 15;
  const int col  = wave * 16 + l15;

  v8f c = {};
  if (bias) {
    float bv = bias[col];
#pragma unroll
    for (int v = 0; v < 8; ++v) c[v] = bv;
  }

#pragma unroll
  for (int kc = 0; kc < 4; ++kc) {
    // A fragment: two aligned 16B LDS reads (k = kc*32 + hi*16 + 0..15)
    const h2* rowp = &xs[l15 * 68 + kc * 16 + hi * 8];
    v8h alo = *(const v8h*)(rowp);
    v8h ahi = *(const v8h*)(rowp + 4);
    v16h a  = __builtin_shufflevector(alo, ahi, 0, 1, 2, 3, 4, 5, 6, 7,
                                      8, 9, 10, 11, 12, 13, 14, 15);
    // B fragment: one aligned 32B global load from packed weights
    v16h b = *(const v16h*)(Wp + (size_t)(((kc * 8 + wave) * 32 + lane) * 16));

    c = __builtin_amdgcn_wmma_f32_16x16x32_f16(false, a, false, b,
                                               (short)0, c, false, false);
  }

#pragma unroll
  for (int v = 0; v < 8; ++v) {               // C/D: row = v + hi*8
    Y[(size_t)(rowBase + v + hi * 8) * GNN_DOUT + col] = c[v];
  }
}

// ------------------------------ edge kernel --------------------------------
// Per 16-edge tile (one wave):
//   C = T[src] gather + bias; C += Xe_tile @ W_edge (WMMA, K=32);
//   ReLU; atomicAdd into H[dst].
__global__ __launch_bounds__(256) void gnn_edge_kernel(
    const float*    __restrict__ Xe,   // [E,32] f32
    const int*      __restrict__ src,  // [E]
    const int*      __restrict__ dst,  // [E]
    const float*    __restrict__ T,    // [Nsrc,128] pre-projected src features
    const _Float16* __restrict__ Wp,   // packed fragments: [1][8][32][16]
    const float*    __restrict__ bias, // [128]
    float*          __restrict__ H,    // [Ndst,128] accumulated with atomics
    int nEdges)
{
  const int lane = threadIdx.x & 31;
  const int wave = threadIdx.x >> 5;
  const int hi   = lane >> 4;
  const int l15  = lane & 15;

  // Preload all 8 B fragments (one 32B load each) and the bias.
  v16h bfrag[8];
#pragma unroll
  for (int g = 0; g < 8; ++g)
    bfrag[g] = *(const v16h*)(Wp + (size_t)((g * 32 + lane) * 16));
  float bcol[8];
#pragma unroll
  for (int g = 0; g < 8; ++g) bcol[g] = bias[g * 16 + l15];

  const int nTiles = nEdges >> 4;             // 16 edges per tile (E % 16 == 0)
  const int step   = gridDim.x * 8;
  for (int tile = blockIdx.x * 8 + wave; tile < nTiles; tile += step) {
    const int e0 = tile << 4;

    // A fragment: 16 edges x K=32 edge features, f32 -> f16 (vectorized loads).
    v16h a;
#pragma unroll
    for (int i = 0; i < 16; ++i) {
      int k = hi * 16 + i;
      a[i] = (_Float16)Xe[(size_t)(e0 + l15) * GNN_DEDGE + k];
    }

    // Edge indices for the 8 C-rows this lane owns (row = v + hi*8).
    int srow[8], drow[8];
#pragma unroll
    for (int v = 0; v < 8; ++v) {
      int e = e0 + v + hi * 8;
      srow[v] = src[e];
      drow[v] = dst[e];
    }

#pragma unroll
    for (int g = 0; g < 8; ++g) {
      const int col = g * 16 + l15;
      v8f c;
#pragma unroll
      for (int v = 0; v < 8; ++v)            // gather pre-projected src + bias
        c[v] = T[(size_t)srow[v] * GNN_DOUT + col] + bcol[g];

      c = __builtin_amdgcn_wmma_f32_16x16x32_f16(false, a, false, bfrag[g],
                                                 (short)0, c, false, false);

#pragma unroll
      for (int v = 0; v < 8; ++v) {          // ReLU + scatter-sum
        float m = c[v] > 0.0f ? c[v] : 0.0f;
        unsafeAtomicAdd(&H[(size_t)drow[v] * GNN_DOUT + col], m);
      }
    }
  }
}

// ------------------------------- launcher ----------------------------------
extern "C" void kernel_launch(void* const* d_in, const int* in_sizes, int n_in,
                              void* d_out, int out_size, void* d_ws, size_t ws_size,
                              hipStream_t stream) {
  (void)in_sizes; (void)n_in; (void)out_size; (void)ws_size;

  const float* X_user    = (const float*)d_in[0];
  const float* X_item    = (const float*)d_in[1];
  const float* Xe_ui     = (const float*)d_in[2];
  const float* Xe_iu     = (const float*)d_in[3];
  const float* W_src_ui  = (const float*)d_in[4];
  const float* W_edge_ui = (const float*)d_in[5];
  const float* b_ui      = (const float*)d_in[6];
  const float* W_src_iu  = (const float*)d_in[7];
  const float* W_edge_iu = (const float*)d_in[8];
  const float* b_iu      = (const float*)d_in[9];
  const float* Wl_user   = (const float*)d_in[10];
  const float* bl_user   = (const float*)d_in[11];
  const float* Wl_item   = (const float*)d_in[12];
  const float* bl_item   = (const float*)d_in[13];
  const int*   src_ui    = (const int*)d_in[14];
  const int*   dst_ui    = (const int*)d_in[15];
  const int*   src_iu    = (const int*)d_in[16];
  const int*   dst_iu    = (const int*)d_in[17];

  float* H_user = (float*)d_out;                               // [100000,128]
  float* H_item = (float*)d_out + (size_t)GNN_N_USER * GNN_DOUT;

  // Workspace layout: packed f16 weights (first 1 MB), then the T buffers.
  char* ws = (char*)d_ws;
  _Float16* WpSrcUi  = (_Float16*)ws;
  _Float16* WpEdgeUi = WpSrcUi  + GNN_DIN   * GNN_DOUT;
  _Float16* WpSrcIu  = WpEdgeUi + GNN_DEDGE * GNN_DOUT;
  _Float16* WpEdgeIu = WpSrcIu  + GNN_DIN   * GNN_DOUT;
  _Float16* WpLUser  = WpEdgeIu + GNN_DEDGE * GNN_DOUT;
  _Float16* WpLItem  = WpLUser  + GNN_DIN   * GNN_DOUT;
  float* T_ui = (float*)(ws + (1 << 20));                      // X_user @ W_src_ui
  float* T_iu = T_ui + (size_t)GNN_N_USER * GNN_DOUT;          // X_item @ W_src_iu

  // 1) Pack weights into WMMA B-fragment order (f16).
  const int nW = GNN_DIN * GNN_DOUT;      // 16384
  const int nE = GNN_DEDGE * GNN_DOUT;    // 4096
  gnn_pack_w<<<(nW + 255) / 256, 256, 0, stream>>>(W_src_ui,  WpSrcUi,  GNN_DIN);
  gnn_pack_w<<<(nE + 255) / 256, 256, 0, stream>>>(W_edge_ui, WpEdgeUi, GNN_DEDGE);
  gnn_pack_w<<<(nW + 255) / 256, 256, 0, stream>>>(W_src_iu,  WpSrcIu,  GNN_DIN);
  gnn_pack_w<<<(nE + 255) / 256, 256, 0, stream>>>(W_edge_iu, WpEdgeIu, GNN_DEDGE);
  gnn_pack_w<<<(nW + 255) / 256, 256, 0, stream>>>(Wl_user,   WpLUser,  GNN_DIN);
  gnn_pack_w<<<(nW + 255) / 256, 256, 0, stream>>>(Wl_item,   WpLItem,  GNN_DIN);

  // 2) Dense WMMA GEMMs: pre-projections (no bias) + self-loop into d_out.
  const int gUser = GNN_N_USER / 16;  // 6250 (exact)
  const int gItem = GNN_N_ITEM / 16;  // 6250 (exact)
  gnn_node_gemm<<<gUser, 256, 0, stream>>>(X_user, WpSrcUi, nullptr, T_ui);
  gnn_node_gemm<<<gItem, 256, 0, stream>>>(X_item, WpSrcIu, nullptr, T_iu);
  gnn_node_gemm<<<gUser, 256, 0, stream>>>(X_user, WpLUser, bl_user, H_user);
  gnn_node_gemm<<<gItem, 256, 0, stream>>>(X_item, WpLItem, bl_item, H_item);

  // 3) Edge MLP + gather + ReLU + atomic scatter-sum.
  const int edgeBlocks = 2048;        // 16384 waves over 50000 tiles
  gnn_edge_kernel<<<edgeBlocks, 256, 0, stream>>>(
      Xe_ui, src_ui, dst_ui, T_ui, WpEdgeUi, b_ui, H_item, GNN_E);
  gnn_edge_kernel<<<edgeBlocks, 256, 0, stream>>>(
      Xe_iu, src_iu, dst_iu, T_iu, WpEdgeIu, b_iu, H_user, GNN_E);
}
